// WindowAttention_944892805589
// MI455X (gfx1250) — matrix operand order repaired and checked by reading, not compiled, not run on hardware
//
#include <hip/hip_runtime.h>
#include <hip/hip_bf16.h>
#include <math.h>

#define DIMC   384
#define HEADS  12
#define HD     32
#define WS7    7
#define NTOK   49
#define NPAD   64
#define NWIN   64          // (56/7)^2
#define NBLK   2048        // 32 * 64
#define PDIM   32
#define HH     56

typedef __attribute__((ext_vector_type(16))) _Float16 v16h;
typedef __attribute__((ext_vector_type(8)))  _Float16 v8h;
typedef __attribute__((ext_vector_type(8)))  float    v8f;

// ---- workspace layout (in _Float16 elements) ----
#define WQ_OFF 0
#define WQ_CNT (3 * DIMC * DIMC)                 // 442368  qkv_w f16
#define WP_OFF (WQ_OFF + WQ_CNT)
#define WP_CNT (DIMC * DIMC)                     // 147456  proj_w f16
#define BM_OFF (WP_OFF + WP_CNT)
#define BM_CNT (NWIN * HEADS * NTOK * NTOK)      // bias+mask table f16

// ---- LDS layout (in _Float16 elements) ----
#define L_XS   0          // [64][384]  x (f16); reused as attention-out "os"
#define L_QS   24576      // [64][384]  Q (pre-scaled)
#define L_KS   49152      // [64][384]  K
#define L_VT   73728      // [384][64]  V transposed [dim][token]
#define L_PS   98304      // [64][64]   scores / probs (in-place softmax)
#define L_XF32 102400     // [49*384] fp32 async staging (75264 B)
#define LDS_HALFS (L_XF32 + NTOK * DIMC * 2)     // 140032 halfs = 280064 B

// A-fragment (16x32 f16): lane holds row (lane&15); K pattern:
//   a[0..15] = A[row][half*16 + 0..15]
__device__ __forceinline__ v16h load_a_frag(const _Float16* base, int stride, int lane) {
  const int row  = lane & 15;
  const int half = lane >> 4;
  const _Float16* p = base + row * stride + half * 16;
  v8h lo = *(const v8h*)(p);
  v8h hi = *(const v8h*)(p + 8);
  v16h r;
#pragma unroll
  for (int i = 0; i < 8; ++i) { r[i] = lo[i]; r[8 + i] = hi[i]; }
  return r;
}

// B-fragment (32x16 f16) from a transposed (Bt = [N][K] row-major) buffer:
//   lane holds column (lane&15); b[i] = Bt[col][half*16 + i], i = 0..15
__device__ __forceinline__ v16h load_b_frag(const _Float16* base, int stride, int lane) {
  const int col  = lane & 15;
  const int half = lane >> 4;
  return *(const v16h*)(base + col * stride + half * 16);
}

__device__ __forceinline__ v8f wmma_f16(v16h a, v16h b, v8f c) {
  return __builtin_amdgcn_wmma_f32_16x16x32_f16(false, a, false, b, (short)0, c, false, false);
}

// ---------------------------------------------------------------------------
// Kernel 1: convert qkv_w and proj_w (fp32, row-major [out][in]) to f16 in ws.
// ---------------------------------------------------------------------------
__global__ void convert_weights(const float* __restrict__ qkv_w,
                                const float* __restrict__ proj_w,
                                _Float16* __restrict__ wdst) {
  const int total = WQ_CNT + WP_CNT;
  for (int i = blockIdx.x * blockDim.x + threadIdx.x; i < total;
       i += gridDim.x * blockDim.x) {
    float v = (i < WQ_CNT) ? qkv_w[i] : proj_w[i - WQ_CNT];
    wdst[i] = (_Float16)v;
  }
}

// ---------------------------------------------------------------------------
// Kernel 2: per-window relative-position MLP bias + shift mask -> f16 table
//   bm[w][h][i][j] = pos_mlp(diff(j,i)) + mask[w][i][j]
// ---------------------------------------------------------------------------
__global__ __launch_bounds__(256)
void build_bias(const float* __restrict__ pos,   // [1,56,56,2]
                const float* __restrict__ mask,  // [64,49,49]
                const float* __restrict__ w1, const float* __restrict__ b1,
                const float* __restrict__ w2, const float* __restrict__ b2,
                _Float16* __restrict__ bmo) {
  const int w  = blockIdx.x;
  const int wy = w >> 3, wx = w & 7;
  for (int pr = threadIdx.x; pr < NTOK * NTOK; pr += blockDim.x) {
    const int i = pr / NTOK, j = pr % NTOK;
    const int iy = i / WS7, ix = i % WS7;
    const int jy = j / WS7, jx = j % WS7;
    const int gi = (wy * WS7 + iy) * HH + (wx * WS7 + ix);
    const int gj = (wy * WS7 + jy) * HH + (wx * WS7 + jx);
    float dy = pos[gj * 2 + 0] - pos[gi * 2 + 0];
    float dx = pos[gj * 2 + 1] - pos[gi * 2 + 1];
    dy = (dy > 0.f ? 1.f : (dy < 0.f ? -1.f : 0.f)) * log1pf(fabsf(dy));
    dx = (dx > 0.f ? 1.f : (dx < 0.f ? -1.f : 0.f)) * log1pf(fabsf(dx));
    float h1[PDIM];
#pragma unroll
    for (int p = 0; p < PDIM; ++p) {
      float v = dy * w1[2 * p + 0] + dx * w1[2 * p + 1] + b1[p];
      h1[p] = v > 0.f ? v : 0.f;
    }
    const float m = mask[(w * NTOK + i) * NTOK + j];
    for (int h = 0; h < HEADS; ++h) {
      float acc = b2[h];
#pragma unroll
      for (int p = 0; p < PDIM; ++p) acc += h1[p] * w2[h * PDIM + p];
      bmo[((size_t)(w * HEADS + h) * NTOK + i) * NTOK + j] = (_Float16)(acc + m);
    }
  }
}

// ---------------------------------------------------------------------------
// Kernel 3: fused window attention. One block = one window (49 tokens).
// x tile staged via GLOBAL_LOAD_ASYNC_TO_LDS (ASYNCcnt), all intermediates
// (x, Q, K, V^T, probs, attn-out) resident in ~274 KB of LDS.
// ---------------------------------------------------------------------------
__global__ __launch_bounds__(256)
void swin_attn_fused(const float* __restrict__ x,
                     const float* __restrict__ qkv_b,
                     const float* __restrict__ proj_b,
                     const _Float16* __restrict__ wq,   // [1152][384] f16
                     const _Float16* __restrict__ wp,   // [384][384]  f16
                     const _Float16* __restrict__ bm,   // [64][12][49][49] f16
                     float* __restrict__ out) {
  extern __shared__ _Float16 smem[];
  _Float16* xs = smem + L_XS;   // x tile; later reused as attention-out
  _Float16* qs = smem + L_QS;
  _Float16* ks = smem + L_KS;
  _Float16* vt = smem + L_VT;   // [384 dim][64 tok]
  _Float16* ps = smem + L_PS;   // [64][64]
  float*    xf = (float*)(smem + L_XF32);
  _Float16* os = xs;            // overlay: x dead after QKV GEMM

  const int tid  = threadIdx.x;
  const int wave = tid >> 5;
  const int lane = tid & 31;
  const int half = lane >> 4;
  const int lcol = lane & 15;
  const int blk  = blockIdx.x;
  const int win  = blk & (NWIN - 1);
  const float qscale = 0.17677669529663687f;  // 1/sqrt(32)

  // ---- phase 0a: async-copy x tile (fp32) directly into LDS staging ----
  {
    const unsigned long long gx =
        (unsigned long long)(uintptr_t)(const void*)(x + (size_t)blk * NTOK * DIMC);
    const unsigned int lds0 = (unsigned int)(uintptr_t)(void*)xf;
    for (int idx = tid; idx < (NTOK * DIMC) / 4; idx += 256) {  // 4704 x 16B
      unsigned long long ga = gx + (unsigned long long)idx * 16ull;
      unsigned int       la = lds0 + (unsigned int)idx * 16u;
      asm volatile("global_load_async_to_lds_b128 %0, %1, off"
                   :: "v"(la), "v"(ga) : "memory");
    }
    asm volatile("s_wait_asynccnt 0x0" ::: "memory");
  }
  __syncthreads();

  // ---- phase 0b: convert staging fp32 -> f16, zero-pad rows 49..63 ----
  for (int idx = tid; idx < NPAD * DIMC; idx += 256)
    xs[idx] = (idx < NTOK * DIMC) ? (_Float16)xf[idx] : (_Float16)0.f;
  __syncthreads();

  // ---- phase 1: QKV = x @ qkv_w^T + b   ([64x384] @ [384x1152]) ----
  // Each wave owns a fixed M-tile (m = wave&3): cache its 12 A-fragments
  // in registers, then sweep 36 column tiles with global-B WMMAs.
  {
    const int m = wave & 3;
    v16h afr[12];
#pragma unroll
    for (int k = 0; k < 12; ++k)
      afr[k] = load_a_frag(xs + m * 16 * DIMC + k * 32, DIMC, lane);

    for (int t = wave; t < 4 * 72; t += 8) {
      const int n = t >> 2;                    // 0..71 (16-wide column tile)
      const _Float16* wrow = wq + (size_t)n * 16 * DIMC;
      __builtin_prefetch(wrow + ((size_t)(t + 8) >> 2) * 0 , 0, 3);  // keep hot
      const float bias = qkv_b[n * 16 + lcol];
      v8f c;
#pragma unroll
      for (int j = 0; j < 8; ++j) c[j] = bias;
#pragma unroll
      for (int k = 0; k < 12; ++k) {
        v16h b = load_b_frag(wrow + k * 32, DIMC, lane);
        c = wmma_f16(afr[k], b, c);
      }
      const int s  = n / 24;                   // 0:Q 1:K 2:V
      const int cc = (n - s * 24) * 16 + lcol; // column within Q/K/V (0..383)
      if (s == 0) {
#pragma unroll
        for (int j = 0; j < 8; ++j)
          qs[(m * 16 + j + half * 8) * DIMC + cc] = (_Float16)(c[j] * qscale);
      } else if (s == 1) {
#pragma unroll
        for (int j = 0; j < 8; ++j)
          ks[(m * 16 + j + half * 8) * DIMC + cc] = (_Float16)c[j];
      } else {  // store V transposed: vt[dim][token]
#pragma unroll
        for (int j = 0; j < 8; ++j)
          vt[cc * NPAD + (m * 16 + j + half * 8)] = (_Float16)c[j];
      }
    }
  }
  __syncthreads();

  // ---- phase 2: per-head attention ----
  for (int h = 0; h < HEADS; ++h) {
    // S = Q_h @ K_h^T  (16 tiles of 16x16, K=32 -> one WMMA each).
    // Fixed m per wave: load the Q fragment once, reuse for both N-tiles.
    {
      const int m = wave & 3;
      v16h a = load_a_frag(qs + m * 16 * DIMC + h * HD, DIMC, lane);
#pragma unroll
      for (int i = 0; i < 2; ++i) {
        const int n = (wave >> 2) + i * 2;
        v16h b = load_b_frag(ks + n * 16 * DIMC + h * HD, DIMC, lane);
        v8f c = {};
        c = wmma_f16(a, b, c);
#pragma unroll
        for (int j = 0; j < 8; ++j)
          ps[(m * 16 + j + half * 8) * NPAD + n * 16 + lcol] = (_Float16)c[j];
      }
    }
    __syncthreads();

    // softmax over rows (add bias+mask table; pad cols/rows -> 0)
    if (tid < NPAD) {
      _Float16* row = ps + tid * NPAD;
      if (tid < NTOK) {
        const _Float16* bmr =
            bm + ((size_t)(win * HEADS + h) * NTOK + tid) * NTOK;
        float mx = -1e30f;
        for (int c = 0; c < NTOK; ++c) {
          float v = (float)row[c] + (float)bmr[c];
          row[c] = (_Float16)v;
          mx = fmaxf(mx, v);
        }
        float sum = 0.f;
        for (int c = 0; c < NTOK; ++c) {
          float e = __expf((float)row[c] - mx);
          sum += e;
          row[c] = (_Float16)e;
        }
        const float inv = 1.f / sum;
        for (int c = 0; c < NTOK; ++c) row[c] = (_Float16)((float)row[c] * inv);
        for (int c = NTOK; c < NPAD; ++c) row[c] = (_Float16)0.f;
      } else {
        for (int c = 0; c < NPAD; ++c) row[c] = (_Float16)0.f;
      }
    }
    __syncthreads();

    // O_h = P @ V_h   (8 tiles: 4 M-tiles x 2 N-tiles, K=64 -> 2 WMMAs)
    {
      const int m = wave >> 1, n = wave & 1;
      v8f c = {};
#pragma unroll
      for (int k = 0; k < 2; ++k) {
        v16h a = load_a_frag(ps + m * 16 * NPAD + k * 32, NPAD, lane);
        v16h b = load_b_frag(vt + (h * HD + n * 16) * NPAD + k * 32, NPAD, lane);
        c = wmma_f16(a, b, c);
      }
#pragma unroll
      for (int j = 0; j < 8; ++j)
        os[(m * 16 + j + half * 8) * DIMC + h * HD + n * 16 + lcol] =
            (_Float16)c[j];
    }
    __syncthreads();
  }

  // ---- phase 3: Y = O @ proj_w^T + b, store fp32 rows < 49 ----
  // Same register-cached A-fragment trick (m fixed per wave, 12 N-tiles).
  {
    const int m = wave & 3;
    v16h afr[12];
#pragma unroll
    for (int k = 0; k < 12; ++k)
      afr[k] = load_a_frag(os + m * 16 * DIMC + k * 32, DIMC, lane);

    float* og = out + (size_t)blk * NTOK * DIMC;
    for (int t = wave; t < 4 * 24; t += 8) {
      const int n = t >> 2;
      const _Float16* wrow = wp + (size_t)n * 16 * DIMC;
      const int ncol = n * 16 + lcol;
      const float pb = proj_b[ncol];
      v8f c;
#pragma unroll
      for (int j = 0; j < 8; ++j) c[j] = pb;
#pragma unroll
      for (int k = 0; k < 12; ++k) {
        v16h b = load_b_frag(wrow + k * 32, DIMC, lane);
        c = wmma_f16(afr[k], b, c);
      }
#pragma unroll
      for (int j = 0; j < 8; ++j) {
        const int r = m * 16 + j + half * 8;
        if (r < NTOK) og[(size_t)r * DIMC + ncol] = c[j];
      }
    }
  }
}

// ---------------------------------------------------------------------------
extern "C" void kernel_launch(void* const* d_in, const int* in_sizes, int n_in,
                              void* d_out, int out_size, void* d_ws, size_t ws_size,
                              hipStream_t stream) {
  const float* x      = (const float*)d_in[0];
  const float* pos    = (const float*)d_in[1];
  const float* mask   = (const float*)d_in[2];
  const float* qkv_w  = (const float*)d_in[3];
  const float* qkv_b  = (const float*)d_in[4];
  const float* proj_w = (const float*)d_in[5];
  const float* proj_b = (const float*)d_in[6];
  const float* w1     = (const float*)d_in[7];
  const float* b1     = (const float*)d_in[8];
  const float* w2     = (const float*)d_in[9];
  const float* b2     = (const float*)d_in[10];

  _Float16* ws = (_Float16*)d_ws;

  convert_weights<<<512, 256, 0, stream>>>(qkv_w, proj_w, ws + WQ_OFF);
  build_bias<<<NWIN, 256, 0, stream>>>(pos, mask, w1, b1, w2, b2, ws + BM_OFF);

  const size_t lds_bytes = (size_t)LDS_HALFS * sizeof(_Float16);  // 280064 B
  swin_attn_fused<<<NBLK, 256, lds_bytes, stream>>>(
      x, qkv_b, proj_b, ws + WQ_OFF, ws + WP_OFF, ws + BM_OFF, (float*)d_out);
}
